// DL4MT_78340203479569
// MI455X (gfx1250) — matrix operand order repaired and checked by reading, not compiled, not run on hardware
//
#include <hip/hip_runtime.h>
#include <hip/hip_bf16.h>
#include <math.h>

typedef __attribute__((ext_vector_type(16))) _Float16 v16h;
typedef __attribute__((ext_vector_type(8)))  _Float16 v8h;
typedef __attribute__((ext_vector_type(8)))  float    v8f;

#define FLAG_BIAS  1
#define FLAG_ACCUM 2
#define FLAG_TANH  4

// ---------------------------------------------------------------------------
// WMMA GEMM: C[M,N] (f32) = A[M,K](f16 row-major) * Bpk (f16, fragment-swizzled
// from W^T), optional +bias[n], +=C, tanh, and f16 copy of C.
// Requirements: M % 32 == 0 (or M==32), N % 64 == 0, K % 32 == 0.
// Block: 256 threads = 8 waves; each block computes a 32(M) x 64(N) region,
// each wave one 16x16 WMMA tile accumulated over K in chunks of 32.
// Bpk layout: ((nt*(K/32) + kc)*32 + lane)*16 + e  where
//   n = nt*16 + (lane&15),  k = kc*32 + (lane>>4)*16 + e   (contiguous k order, same as A)
// ---------------------------------------------------------------------------
__global__ void gemm_wmma(const _Float16* __restrict__ A,
                          const _Float16* __restrict__ Bpk,
                          const float* __restrict__ bias,
                          float* __restrict__ C, _Float16* __restrict__ C16,
                          int M, int N, int K, int flags) {
  const int w    = threadIdx.x >> 5;     // 0..7
  const int lane = threadIdx.x & 31;
  const int half = lane >> 4;            // 0/1
  const int l15  = lane & 15;
  const int wm = w & 1, wn = w >> 1;     // 2 (M) x 4 (N) wave grid
  const int m0 = blockIdx.y * 32 + wm * 16;
  const int n0 = blockIdx.x * 64 + wn * 16;
  const int Kc = K >> 5;

  const _Float16* Arow = A + (size_t)(m0 + l15) * K;
  const _Float16* bptr = Bpk + (((size_t)(n0 >> 4) * Kc) * 32 + lane) * 16;

  v8f acc = {};
  for (int k0 = 0; k0 < K; k0 += 32) {
    // A fragment (contiguous K order, matches Bpk):
    //   elems 0..15 -> K = k0 + half*16 .. +15
    v8h alo = *(const v8h*)(Arow + k0 + half * 16);
    v8h ahi = *(const v8h*)(Arow + k0 + half * 16 + 8);
    v16h a;
#pragma unroll
    for (int i = 0; i < 8; ++i) { a[i] = alo[i]; a[8 + i] = ahi[i]; }
    // B fragment: one contiguous 32B load from the swizzled pack
    v16h b = *(const v16h*)bptr;
    __builtin_prefetch((const void*)(bptr + 512), 0, 1);  // next K-chunk
    bptr += 512;
    acc = __builtin_amdgcn_wmma_f32_16x16x32_f16(
        false, a, false, b, (short)0, acc, false, false);
  }

  // C/D layout: VGPR r -> M = m0 + r + 8*half, N = n0 + l15
#pragma unroll
  for (int r = 0; r < 8; ++r) {
    int m = m0 + r + half * 8;
    int n = n0 + l15;
    size_t idx = (size_t)m * N + n;
    float v = acc[r];
    if (flags & FLAG_BIAS)  v += bias[n];
    if (flags & FLAG_ACCUM) v += C[idx];
    if (flags & FLAG_TANH)  v  = tanhf(v);
    C[idx] = v;
    if (C16) C16[idx] = (_Float16)v;
  }
}

// Pack W[Nsrc,Ksrc] (row-major, f32) into f16 B-fragment order for X @ W^T.
__global__ void pack_bT(const float* __restrict__ W, _Float16* __restrict__ dst,
                        int Ksrc, int Nsrc) {
  long tid = (long)blockIdx.x * blockDim.x + threadIdx.x;
  long total = (long)Ksrc * Nsrc;
  if (tid >= total) return;
  int e    = (int)(tid & 15);
  int lane = (int)((tid >> 4) & 31);
  long rest = tid >> 9;
  int Kc = Ksrc >> 5;
  int kc = (int)(rest % Kc);
  int nt = (int)(rest / Kc);
  int k = kc * 32 + (lane >> 4) * 16 + e;
  int n = nt * 16 + (lane & 15);
  dst[tid] = (_Float16)W[(size_t)n * Ksrc + k];
}

// Scaled concat pack of the six mix projection weights into [3712,512] fragments.
__global__ void pack_mix_b(const float* Wse, const float* Wl0, const float* Wl1,
                           const float* Wl2, const float* Wdep, const float* Whead,
                           const float* __restrict__ sw, _Float16* __restrict__ dst) {
  const int Kpad = 3712, Npad = 512, Kc = Kpad >> 5;
  long tid = (long)blockIdx.x * blockDim.x + threadIdx.x;
  if (tid >= (long)Kpad * Npad) return;
  int e    = (int)(tid & 15);
  int lane = (int)((tid >> 4) & 31);
  long rest = tid >> 9;
  int kc = (int)(rest % Kc);
  int nt = (int)(rest / Kc);
  int k = kc * 32 + (lane >> 4) * 16 + e;
  int n = nt * 16 + (lane & 15);
  float v = 0.f;
  if      (k < 100)  v = Wse[(size_t)n * 100 + k]          * sw[0];
  else if (k < 900)  v = Wl0[(size_t)n * 800 + (k - 100)]  * sw[1];
  else if (k < 1700) v = Wl1[(size_t)n * 800 + (k - 900)]  * sw[2];
  else if (k < 2500) v = Wl2[(size_t)n * 800 + (k - 1700)] * sw[3];
  else if (k < 3100) v = Wdep[(size_t)n * 600 + (k - 2500)] * sw[4];
  else if (k < 3700) v = Whead[(size_t)n * 600 + (k - 3100)] * sw[5];
  dst[tid] = (_Float16)v;
}

// Concat syntax features into mix A [1280 (b*S+s), 3712] f16.
__global__ void pack_mix_a(const float* se, const float* l0, const float* l1,
                           const float* l2, const float* dep, const float* head,
                           _Float16* __restrict__ dst) {
  long tid = (long)blockIdx.x * blockDim.x + threadIdx.x;
  if (tid >= (long)1280 * 3712) return;
  long row = tid / 3712;
  int  k   = (int)(tid % 3712);
  float v = 0.f;
  if      (k < 100)  v = se [row * 100 + k];
  else if (k < 900)  v = l0 [row * 800 + (k - 100)];
  else if (k < 1700) v = l1 [row * 800 + (k - 900)];
  else if (k < 2500) v = l2 [row * 800 + (k - 1700)];
  else if (k < 3100) v = dep[row * 600 + (k - 2500)];
  else if (k < 3700) v = head[row * 600 + (k - 3100)];
  dst[tid] = (_Float16)v;
}

// softmax(mix_w) * gamma -> sw[6]
__global__ void mixw_k(const float* __restrict__ w, const float* __restrict__ gamma,
                       float* __restrict__ sw) {
  if (threadIdx.x == 0 && blockIdx.x == 0) {
    float m = w[0];
    for (int i = 1; i < 6; ++i) m = fmaxf(m, w[i]);
    float e[6], s = 0.f;
    for (int i = 0; i < 6; ++i) { e[i] = expf(w[i] - m); s += e[i]; }
    for (int i = 0; i < 6; ++i) sw[i] = gamma[0] * e[i] / s;
  }
}

// enc_in (xseq) pack: rows s*B+b, cols [emb(512) | synmix(512)], f16.
__global__ void pack_enc_in(const int* __restrict__ xs, const float* __restrict__ src_emb,
                            const float* __restrict__ synmix, _Float16* __restrict__ dst,
                            int B, int S) {
  long tid = (long)blockIdx.x * blockDim.x + threadIdx.x;
  if (tid >= (long)B * S * 1024) return;
  long row = tid >> 10;             // s*B + b
  int  e   = (int)(tid & 1023);
  int  s = (int)(row / B), b = (int)(row % B);
  float v;
  if (e < 512) v = src_emb[(size_t)xs[b * S + s] * 512 + e];
  else         v = synmix[((size_t)b * S + s) * 512 + (e - 512)];
  dst[tid] = (_Float16)v;
}

// yemb pack: rows t*B+b, 512 cols, f16.
__global__ void pack_yemb(const int* __restrict__ y, const float* __restrict__ tgt_emb,
                          _Float16* __restrict__ dst, int B, int T) {
  long tid = (long)blockIdx.x * blockDim.x + threadIdx.x;
  if (tid >= (long)B * T * 512) return;
  long row = tid >> 9;              // t*B + b
  int  e   = (int)(tid & 511);
  dst[tid] = (_Float16)tgt_emb[(size_t)y[row] * 512 + e];
}

__global__ void zero_h(float* h, _Float16* h16, int n) {
  int i = blockIdx.x * blockDim.x + threadIdx.x;
  if (i < n) { h[i] = 0.f; h16[i] = (_Float16)0.f; }
}

// GRU gate fusion: gates (r,z,n); writes h (f32+f16) and optional strided copy.
__global__ void gru_gate(const float* __restrict__ gi, const float* __restrict__ gh,
                         const float* __restrict__ hprev,
                         float* __restrict__ hout, _Float16* __restrict__ hout16,
                         float* __restrict__ extra, _Float16* __restrict__ extra16,
                         long extra_stride, int B, int H) {
  int i = blockIdx.x * blockDim.x + threadIdx.x;
  if (i >= B * H) return;
  int b = i / H, j = i % H;
  size_t base = (size_t)b * 3 * H + j;
  float r = 1.f / (1.f + expf(-(gi[base]         + gh[base])));
  float z = 1.f / (1.f + expf(-(gi[base + H]     + gh[base + H])));
  float n = tanhf(gi[base + 2 * H] + r * gh[base + 2 * H]);
  float h = (1.f - z) * n + z * hprev[i];
  hout[i] = h; hout16[i] = (_Float16)h;
  if (extra) {
    size_t e = (size_t)b * extra_stride + j;
    extra[e] = h; extra16[e] = (_Float16)h;
  }
}

// masked mean of ctx over S -> f16 [B,2H]
__global__ void ctx_mean_k(const float* __restrict__ ctx, const int* __restrict__ xs,
                           _Float16* __restrict__ out16, int B, int S, int C2) {
  int i = blockIdx.x * blockDim.x + threadIdx.x;
  if (i >= B * C2) return;
  int b = i / C2, c = i % C2;
  float s = 0.f, cnt = 0.f;
  for (int t = 0; t < S; ++t) {
    float np = (xs[b * S + t] == 0) ? 0.f : 1.f;
    s += np * ctx[((size_t)b * S + t) * C2 + c];
    cnt += np;
  }
  out16[i] = (_Float16)(s / cnt);
}

// Bahdanau scores: one wave per (b,s).
__global__ void attn_scores(const float* __restrict__ q, const float* __restrict__ cache,
                            const float* __restrict__ vvec, const int* __restrict__ xs,
                            float* __restrict__ scores, int B, int S, int H) {
  int wave = (int)((blockIdx.x * blockDim.x + threadIdx.x) >> 5);
  int lane = threadIdx.x & 31;
  if (wave >= B * S) return;
  int b = wave / S, s = wave % S;
  const float* qb = q + (size_t)b * H;
  const float* cb = cache + (size_t)wave * H;
  float sum = 0.f;
  for (int h = lane; h < H; h += 32) sum += tanhf(qb[h] + cb[h]) * vvec[h];
  for (int off = 16; off; off >>= 1) sum += __shfl_down(sum, off, 32);
  if (lane == 0) scores[wave] = (xs[b * S + s] == 0) ? -1e9f : sum;
}

__global__ void attn_softmax(const float* __restrict__ scores, float* __restrict__ aw,
                             int S) {
  int b = blockIdx.x, lane = threadIdx.x;           // blockDim = 32
  float v1 = (lane < S)      ? scores[b * S + lane]      : -3.0e38f;
  float v2 = (lane + 32 < S) ? scores[b * S + lane + 32] : -3.0e38f;
  float m = fmaxf(v1, v2);
  for (int off = 16; off; off >>= 1) m = fmaxf(m, __shfl_xor(m, off, 32));
  float e1 = (lane < S)      ? expf(v1 - m) : 0.f;
  float e2 = (lane + 32 < S) ? expf(v2 - m) : 0.f;
  float s = e1 + e2;
  for (int off = 16; off; off >>= 1) s += __shfl_xor(s, off, 32);
  if (lane < S)      aw[b * S + lane]      = e1 / s;
  if (lane + 32 < S) aw[b * S + lane + 32] = e2 / s;
}

// av[b,c] = sum_s aw[b,s]*ctx[b,s,c]; writes f32 + f16 slice of attns.
__global__ void attn_av(const float* __restrict__ aw, const float* __restrict__ ctx,
                        float* __restrict__ att, _Float16* __restrict__ att16,
                        int B, int S, int C2) {
  int i = blockIdx.x * blockDim.x + threadIdx.x;
  if (i >= B * C2) return;
  int b = i / C2, c = i % C2;
  float s = 0.f;
  for (int t = 0; t < S; ++t) s += aw[b * S + t] * ctx[((size_t)b * S + t) * C2 + c];
  att[i] = s; att16[i] = (_Float16)s;
}

// row-wise log_softmax with PAD-column masking + [T,B]->[B,T] transpose on write.
__global__ void logsoftmax_k(const float* __restrict__ g, float* __restrict__ out,
                             int B, int T, int V) {
  __shared__ float red[8];
  int r = blockIdx.x;                     // t*B + b
  int t = r / B, b = r % B;
  const float* row = g + (size_t)r * V;
  int w = threadIdx.x >> 5, lane = threadIdx.x & 31;
  float m = -3.0e38f;
  for (int v = threadIdx.x; v < V; v += blockDim.x) {
    float x = (v == 0) ? -1e9f : row[v];
    m = fmaxf(m, x);
  }
  for (int off = 16; off; off >>= 1) m = fmaxf(m, __shfl_xor(m, off, 32));
  if (lane == 0) red[w] = m;
  __syncthreads();
  float m0 = red[0];
#pragma unroll
  for (int i = 1; i < 8; ++i) m0 = fmaxf(m0, red[i]);
  __syncthreads();
  float s = 0.f;
  for (int v = threadIdx.x; v < V; v += blockDim.x) {
    float x = (v == 0) ? -1e9f : row[v];
    s += expf(x - m0);
  }
  for (int off = 16; off; off >>= 1) s += __shfl_xor(s, off, 32);
  if (lane == 0) red[w] = s;
  __syncthreads();
  float stot = 0.f;
#pragma unroll
  for (int i = 0; i < 8; ++i) stot += red[i];
  float lse = m0 + logf(stot);
  float* orow = out + ((size_t)b * T + t) * V;
  for (int v = threadIdx.x; v < V; v += blockDim.x) {
    float x = (v == 0) ? -1e9f : row[v];
    orow[v] = x - lse;
  }
}

static inline long cdivl(long a, long b) { return (a + b - 1) / b; }

extern "C" void kernel_launch(void* const* d_in, const int* in_sizes, int n_in,
                              void* d_out, int out_size, void* d_ws, size_t ws_size,
                              hipStream_t stream) {
  const int B = 32, S = 40, T = 40, E = 512, H = 1024, VT = 32000;
  const int H2 = 2048, H3 = 3072;
  (void)in_sizes; (void)n_in; (void)out_size; (void)ws_size;

  // ---- inputs (setup_inputs order) ----
  const int*   xs      = (const int*)d_in[0];
  const int*   y       = (const int*)d_in[1];
  const float* syn_emb = (const float*)d_in[3];
  const float* syn_l0  = (const float*)d_in[4];
  const float* syn_l1  = (const float*)d_in[5];
  const float* syn_l2  = (const float*)d_in[6];
  const float* syn_dep = (const float*)d_in[7];
  const float* syn_head= (const float*)d_in[8];
  const float* src_emb = (const float*)d_in[9];
  const float* tgt_emb = (const float*)d_in[10];
  const float* W_semb  = (const float*)d_in[11];
  const float* W_sl0   = (const float*)d_in[12];
  const float* W_sl1   = (const float*)d_in[13];
  const float* W_sl2   = (const float*)d_in[14];
  const float* W_sdep  = (const float*)d_in[15];
  const float* W_shead = (const float*)d_in[16];
  const float* mix_w   = (const float*)d_in[17];
  const float* mix_g   = (const float*)d_in[18];
  const float* ef_Wih  = (const float*)d_in[19];
  const float* ef_Whh  = (const float*)d_in[20];
  const float* ef_bih  = (const float*)d_in[21];
  const float* ef_bhh  = (const float*)d_in[22];
  const float* eb_Wih  = (const float*)d_in[23];
  const float* eb_Whh  = (const float*)d_in[24];
  const float* eb_bih  = (const float*)d_in[25];
  const float* eb_bhh  = (const float*)d_in[26];
  const float* br_W    = (const float*)d_in[27];
  const float* br_b    = (const float*)d_in[28];
  const float* g1_Wih  = (const float*)d_in[29];
  const float* g1_Whh  = (const float*)d_in[30];
  const float* g1_bih  = (const float*)d_in[31];
  const float* g1_bhh  = (const float*)d_in[32];
  const float* att_Wq  = (const float*)d_in[33];
  const float* att_bq  = (const float*)d_in[34];
  const float* att_Wk  = (const float*)d_in[35];
  const float* att_v   = (const float*)d_in[36];
  const float* g2_Wih  = (const float*)d_in[37];
  const float* g2_Whh  = (const float*)d_in[38];
  const float* g2_bih  = (const float*)d_in[39];
  const float* g2_bhh  = (const float*)d_in[40];
  const float* dec_Wi  = (const float*)d_in[41];
  const float* dec_bi  = (const float*)d_in[42];
  const float* dec_Wh  = (const float*)d_in[43];
  const float* dec_bh  = (const float*)d_in[44];
  const float* dec_Wc  = (const float*)d_in[45];
  const float* dec_bc  = (const float*)d_in[46];
  const float* gen_W   = (const float*)d_in[47];
  float* out = (float*)d_out;

  // ---- workspace bump allocator ----
  char* cur = (char*)d_ws;
  auto alloc = [&](size_t bytes) -> char* {
    char* r = cur; cur += (bytes + 255) & ~(size_t)255; return r;
  };
  const long BS = (long)B * S;   // 1280
  const long BT = (long)B * T;   // 1280

  float*     sw      = (float*)alloc(6 * 4);
  _Float16*  mixB    = (_Float16*)alloc((size_t)3712 * 512 * 2);
  _Float16*  mixA    = (_Float16*)alloc((size_t)BS * 3712 * 2);
  float*     synmix  = (float*)alloc((size_t)BS * 512 * 4);
  _Float16*  encA    = (_Float16*)alloc((size_t)BS * 1024 * 2);
  _Float16*  yembA   = (_Float16*)alloc((size_t)BT * 512 * 2);
  _Float16*  pefWih  = (_Float16*)alloc((size_t)1024 * H3 * 2);
  _Float16*  pefWhh  = (_Float16*)alloc((size_t)1024 * H3 * 2);
  _Float16*  pebWih  = (_Float16*)alloc((size_t)1024 * H3 * 2);
  _Float16*  pebWhh  = (_Float16*)alloc((size_t)1024 * H3 * 2);
  _Float16*  pbrW    = (_Float16*)alloc((size_t)H2 * H * 2);
  _Float16*  pattWk  = (_Float16*)alloc((size_t)H2 * H * 2);
  _Float16*  pg1Wih  = (_Float16*)alloc((size_t)E * H3 * 2);
  _Float16*  pg1Whh  = (_Float16*)alloc((size_t)H * H3 * 2);
  _Float16*  pattWq  = (_Float16*)alloc((size_t)H * H * 2);
  _Float16*  pg2Wih  = (_Float16*)alloc((size_t)H2 * H3 * 2);
  _Float16*  pg2Whh  = (_Float16*)alloc((size_t)H * H3 * 2);
  _Float16*  pdecWi  = (_Float16*)alloc((size_t)E * E * 2);
  _Float16*  pdecWh  = (_Float16*)alloc((size_t)H * E * 2);
  _Float16*  pdecWc  = (_Float16*)alloc((size_t)H2 * E * 2);
  _Float16*  pgenW   = (_Float16*)alloc((size_t)E * VT * 2);
  float*     gif     = (float*)alloc((size_t)BS * H3 * 4);
  float*     gib     = (float*)alloc((size_t)BS * H3 * 4);
  float*     gi1     = (float*)alloc((size_t)BT * H3 * 4);
  float*     ghb     = (float*)alloc((size_t)B * H3 * 4);
  float*     gi2     = (float*)alloc((size_t)B * H3 * 4);
  float*     hbuf    = (float*)alloc((size_t)B * H * 4);
  _Float16*  h16     = (_Float16*)alloc((size_t)B * H * 2);
  float*     h1buf   = (float*)alloc((size_t)B * H * 4);
  _Float16*  h1_16   = (_Float16*)alloc((size_t)B * H * 2);
  float*     qbuf    = (float*)alloc((size_t)B * H * 4);
  float*     scores  = (float*)alloc((size_t)BS * 4);
  float*     aw      = (float*)alloc((size_t)BS * 4);
  float*     ctx     = (float*)alloc((size_t)BS * H2 * 4);
  _Float16*  ctx16   = (_Float16*)alloc((size_t)BS * H2 * 2);
  _Float16*  cmean16 = (_Float16*)alloc((size_t)B * H2 * 2);
  float*     cache   = (float*)alloc((size_t)BS * H * 4);
  float*     attns   = (float*)alloc((size_t)BT * H2 * 4);
  _Float16*  attns16 = (_Float16*)alloc((size_t)BT * H2 * 2);
  float*     outs    = (float*)alloc((size_t)BT * H * 4);
  _Float16*  outs16  = (_Float16*)alloc((size_t)BT * H * 2);
  float*     pre     = (float*)alloc((size_t)BT * E * 4);
  _Float16*  lg16    = (_Float16*)alloc((size_t)BT * E * 2);
  float*     gsc     = (float*)alloc((size_t)BT * VT * 4);

  auto gemm = [&](const _Float16* A, const _Float16* Bp, const float* bias,
                  float* C, _Float16* C16, int M, int N, int K, int flags) {
    dim3 g((unsigned)(N / 64), (unsigned)(M / 32));
    gemm_wmma<<<g, 256, 0, stream>>>(A, Bp, bias, C, C16, M, N, K, flags);
  };
  auto packT = [&](const float* W, _Float16* dst, int K, int N) {
    long tot = (long)K * N;
    pack_bT<<<(unsigned)cdivl(tot, 256), 256, 0, stream>>>(W, dst, K, N);
  };

  // ---- weight packing ----
  packT(ef_Wih, pefWih, 1024, H3);  packT(ef_Whh, pefWhh, H, H3);
  packT(eb_Wih, pebWih, 1024, H3);  packT(eb_Whh, pebWhh, H, H3);
  packT(br_W,   pbrW,   H2, H);     packT(att_Wk, pattWk, H2, H);
  packT(g1_Wih, pg1Wih, E, H3);     packT(g1_Whh, pg1Whh, H, H3);
  packT(att_Wq, pattWq, H, H);
  packT(g2_Wih, pg2Wih, H2, H3);    packT(g2_Whh, pg2Whh, H, H3);
  packT(dec_Wi, pdecWi, E, E);      packT(dec_Wh, pdecWh, H, E);
  packT(dec_Wc, pdecWc, H2, E);     packT(gen_W,  pgenW,  E, VT);

  // ---- scalar mix + encoder input ----
  mixw_k<<<1, 32, 0, stream>>>(mix_w, mix_g, sw);
  pack_mix_b<<<(unsigned)cdivl((long)3712 * 512, 256), 256, 0, stream>>>(
      W_semb, W_sl0, W_sl1, W_sl2, W_sdep, W_shead, sw, mixB);
  pack_mix_a<<<(unsigned)cdivl(BS * 3712, 256), 256, 0, stream>>>(
      syn_emb, syn_l0, syn_l1, syn_l2, syn_dep, syn_head, mixA);
  gemm(mixA, mixB, nullptr, synmix, nullptr, (int)BS, 512, 3712, 0);
  pack_enc_in<<<(unsigned)cdivl(BS * 1024, 256), 256, 0, stream>>>(
      xs, src_emb, synmix, encA, B, S);
  pack_yemb<<<(unsigned)cdivl(BT * 512, 256), 256, 0, stream>>>(
      y, tgt_emb, yembA, B, T);

  // ---- batched input-side GRU GEMMs ----
  gemm(encA, pefWih, ef_bih, gif, nullptr, (int)BS, H3, 1024, FLAG_BIAS);
  gemm(encA, pebWih, eb_bih, gib, nullptr, (int)BS, H3, 1024, FLAG_BIAS);
  gemm(yembA, pg1Wih, g1_bih, gi1, nullptr, (int)BT, H3, E, FLAG_BIAS);

  const unsigned gateG = (unsigned)cdivl((long)B * H, 256);
  // ---- encoder forward scan ----
  zero_h<<<gateG, 256, 0, stream>>>(hbuf, h16, B * H);
  for (int s = 0; s < S; ++s) {
    gemm(h16, pefWhh, ef_bhh, ghb, nullptr, B, H3, H, FLAG_BIAS);
    gru_gate<<<gateG, 256, 0, stream>>>(
        gif + (size_t)s * B * H3, ghb, hbuf, hbuf, h16,
        ctx + (size_t)s * H2, ctx16 + (size_t)s * H2, (long)S * H2, B, H);
  }
  // ---- encoder backward scan ----
  zero_h<<<gateG, 256, 0, stream>>>(hbuf, h16, B * H);
  for (int s = 0; s < S; ++s) {
    int so = S - 1 - s;   // original index of this reversed step
    gemm(h16, pebWhh, eb_bhh, ghb, nullptr, B, H3, H, FLAG_BIAS);
    gru_gate<<<gateG, 256, 0, stream>>>(
        gib + (size_t)so * B * H3, ghb, hbuf, hbuf, h16,
        ctx + (size_t)so * H2 + H, ctx16 + (size_t)so * H2 + H, (long)S * H2, B, H);
  }

  // ---- bridge + attention key cache ----
  ctx_mean_k<<<(unsigned)cdivl((long)B * H2, 256), 256, 0, stream>>>(
      ctx, xs, cmean16, B, S, H2);
  gemm(cmean16, pbrW, br_b, hbuf, h16, B, H, H2, FLAG_BIAS | FLAG_TANH);
  gemm(ctx16, pattWk, nullptr, cache, nullptr, (int)BS, H, H2, 0);

  // ---- decoder scan ----
  for (int t = 0; t < T; ++t) {
    gemm(h16, pg1Whh, g1_bhh, ghb, nullptr, B, H3, H, FLAG_BIAS);
    gru_gate<<<gateG, 256, 0, stream>>>(
        gi1 + (size_t)t * B * H3, ghb, hbuf, h1buf, h1_16,
        nullptr, nullptr, 0, B, H);
    gemm(h1_16, pattWq, att_bq, qbuf, nullptr, B, H, H, FLAG_BIAS);
    attn_scores<<<(unsigned)cdivl(BS, 8), 256, 0, stream>>>(
        qbuf, cache, att_v, xs, scores, B, S, H);
    attn_softmax<<<B, 32, 0, stream>>>(scores, aw, S);
    attn_av<<<(unsigned)cdivl((long)B * H2, 256), 256, 0, stream>>>(
        aw, ctx, attns + (size_t)t * B * H2, attns16 + (size_t)t * B * H2, B, S, H2);
    gemm(attns16 + (size_t)t * B * H2, pg2Wih, g2_bih, gi2, nullptr, B, H3, H2, FLAG_BIAS);
    gemm(h1_16, pg2Whh, g2_bhh, ghb, nullptr, B, H3, H, FLAG_BIAS);
    gru_gate<<<gateG, 256, 0, stream>>>(
        gi2, ghb, h1buf, hbuf, h16,
        outs + (size_t)t * B * H, outs16 + (size_t)t * B * H, (long)H, B, H);
  }

  // ---- output head ----
  gemm(yembA,   pdecWi, dec_bi, pre, nullptr, (int)BT, E, E,  FLAG_BIAS);
  gemm(outs16,  pdecWh, dec_bh, pre, nullptr, (int)BT, E, H,  FLAG_BIAS | FLAG_ACCUM);
  gemm(attns16, pdecWc, dec_bc, pre, lg16,    (int)BT, E, H2, FLAG_BIAS | FLAG_ACCUM | FLAG_TANH);
  gemm(lg16, pgenW, nullptr, gsc, nullptr, (int)BT, VT, E, 0);
  logsoftmax_k<<<(unsigned)BT, 256, 0, stream>>>(gsc, out, B, T, VT);
}
